// Decoder_11776800326238
// MI455X (gfx1250) — matrix operand order repaired and checked by reading, not compiled, not run on hardware
//
#include <hip/hip_runtime.h>
#include <cstdint>

// ---------------------------------------------------------------------------
// GRU decoder for MI455X (gfx1250, wave32, WMMA).
// Phases: (1) transpose+bf16 weights, (2) xW GEMM, (3) persistent GRU scan,
// (4) output-projection GEMM (dominant: 268 GFLOP, bf16 WMMA / f32 accum).
// ---------------------------------------------------------------------------

typedef __bf16 bf16_t;
typedef __attribute__((ext_vector_type(16))) __bf16 v16bf;
typedef __attribute__((ext_vector_type(8)))  __bf16 v8bf;
typedef __attribute__((ext_vector_type(4)))  __bf16 v4bf;
typedef __attribute__((ext_vector_type(8)))  float  v8f;

typedef unsigned int tdm_v4u __attribute__((ext_vector_type(4)));
typedef int          tdm_v8i __attribute__((ext_vector_type(8)));
typedef int          tdm_v4i __attribute__((ext_vector_type(4)));

#if __has_builtin(__builtin_amdgcn_tensor_load_to_lds) && __has_builtin(__builtin_amdgcn_s_wait_tensorcnt)
#define HAVE_TDM 1
#else
#define HAVE_TDM 0
#endif

#define BB 16
#define TT 256
#define HH 1024
#define EE 512
#define VV 32000
#define GG 3072          // 3*H
#define MM 4096          // B*T

// workspace layout (bytes)
#define OFF_H    ((size_t)0)                       //  16*1024*4      = 64 KB
#define OFF_XW   ((size_t)65536)                   // 4096*3072*4     = 48 MB
#define OFF_HS   (OFF_XW  + (size_t)4096*3072*4)   // 4096*1024*2     =  8 MB (bf16)
#define OFF_WIT  (OFF_HS  + (size_t)4096*1024*2)   // 3072*512*2
#define OFF_WHT  (OFF_WIT + (size_t)3072*512*2)    // 3072*1024*2
#define OFF_WOT  (OFF_WHT + (size_t)3072*1024*2)   // 32000*1024*2
#define OFF_SYNC (OFF_WOT + (size_t)32000*1024*2)  // 2 * u32

// ---------------------------------------------------------------------------
// helpers
// ---------------------------------------------------------------------------
__device__ __forceinline__ uint32_t lds_off(const void* p) {
    // flat LDS aperture: low 32 bits of a generic LDS pointer are the LDS byte
    // offset within the wave's allocation (ISA 10.2 aperture mapping).
    return (uint32_t)(uintptr_t)p;
}

__device__ __forceinline__ v16bf cat16(v8bf lo, v8bf hi) {
    return __builtin_shufflevector(lo, hi, 0,1,2,3,4,5,6,7,8,9,10,11,12,13,14,15);
}

// A-matrix fragment (16x32 bf16, row-major, row stride = ldk elements).
// Layout per ISA 7.12.2: lanes 0-15: M=lane, K = 0..7 then 16..23;
//                        lanes 16-31: K = 8..15 then 24..31.
__device__ __forceinline__ v16bf load_a_frag(const bf16_t* base, int ldk) {
    const int lane = threadIdx.x & 31;
    const int m    = lane & 15;
    const int kh   = (lane >> 4) * 8;
    const bf16_t* r = base + (size_t)m * ldk;
    v8bf lo = *(const v8bf*)(r + kh);
    v8bf hi = *(const v8bf*)(r + 16 + kh);
    return cat16(lo, hi);
}

// B-matrix fragment (32x16 bf16) from column-major storage [n][k] (row stride
// ldk over k). lanes 0-15: N=lane, K=0..7 then 16..23; lanes 16-31: N=lane-16, K=8..15 then 24..31.
__device__ __forceinline__ v16bf load_b_frag(const bf16_t* base, int ldk) {
    const int lane = threadIdx.x & 31;
    const int n    = lane & 15;
    const int kh   = (lane >> 4) * 8;
    const bf16_t* r = base + (size_t)n * ldk;
    v8bf lo = *(const v8bf*)(r + kh);
    v8bf hi = *(const v8bf*)(r + 16 + kh);
    return cat16(lo, hi);
}

__device__ __forceinline__ v8f wmma_bf16(v16bf a, v16bf b, v8f c) {
    return __builtin_amdgcn_wmma_f32_16x16x32_bf16(false, a, false, b,
                                                   (short)0, c, false, false);
}

__device__ __forceinline__ float sigm(float x) {
    return 1.0f / (1.0f + __expf(-x));
}

#if HAVE_TDM
// Issue a 2D Tensor-Data-Mover load: tile_d1 rows x tile_d0 contiguous bf16
// elements (row stride stride0 elements) from global -> packed LDS.
// Descriptor fields per CDNA5 ISA D# group0/group1.
__device__ __forceinline__ void tdm_load_2d(uint32_t lds_byte, const void* gptr,
                                            uint32_t tile_d0, uint32_t tile_d1,
                                            uint32_t stride0,
                                            uint32_t tdim0, uint32_t tdim1) {
    const uint64_t ga = (uint64_t)(uintptr_t)gptr;
    tdm_v4u g0;
    g0[0] = 1u;                                        // count=1, user-mode
    g0[1] = lds_byte;                                  // lds_addr
    g0[2] = (uint32_t)(ga & 0xFFFFFFFFu);              // global_addr[31:0]
    g0[3] = (uint32_t)((ga >> 32) & 0x1FFFFFFu) | (2u << 30);  // [56:32] | type=2
    tdm_v8i g1;
    g1[0] = (int)(1u << 16);                           // wg_mask=0, data_size=1 (2B)
    g1[1] = (int)((tdim0 & 0xFFFFu) << 16);            // tensor_dim0[15:0]
    g1[2] = (int)(((tdim0 >> 16) & 0xFFFFu) | ((tdim1 & 0xFFFFu) << 16));
    g1[3] = (int)(((tdim1 >> 16) & 0xFFFFu) | ((tile_d0 & 0xFFFFu) << 16));
    g1[4] = (int)(tile_d1 & 0xFFFFu);                  // tile_dim1 (tile_dim2=0)
    g1[5] = (int)stride0;                              // tensor_dim0_stride[31:0]
    g1[6] = 0;
    g1[7] = 0;
    tdm_v4i z4 = {0, 0, 0, 0};
#if defined(__clang_major__) && (__clang_major__ >= 23)
    tdm_v8i z8 = {0, 0, 0, 0, 0, 0, 0, 0};
    __builtin_amdgcn_tensor_load_to_lds(g0, g1, z4, z4, z8, 0);
#else
    __builtin_amdgcn_tensor_load_to_lds(g0, g1, z4, z4, 0);
#endif
}
#endif

// ---------------------------------------------------------------------------
// kernel 0: init sync counters
// ---------------------------------------------------------------------------
__global__ void k_init_sync(unsigned* sync) {
    if (threadIdx.x < 2) sync[threadIdx.x] = 0u;
}

// ---------------------------------------------------------------------------
// kernel 1: tiled transpose fp32 [K][N] -> bf16 [N][K]
// ---------------------------------------------------------------------------
__global__ void k_transpose_bf16(const float* __restrict__ src,
                                 bf16_t* __restrict__ dst, int K, int N) {
    __shared__ float tile[32][33];
    const int kb = blockIdx.y * 32, nb = blockIdx.x * 32;
    for (int i = threadIdx.y; i < 32; i += 8) {
        tile[i][threadIdx.x] = src[(size_t)(kb + i) * N + (nb + threadIdx.x)];
    }
    __syncthreads();
    for (int i = threadIdx.y; i < 32; i += 8) {
        dst[(size_t)(nb + i) * K + (kb + threadIdx.x)] =
            (bf16_t)tile[threadIdx.x][i];
    }
}

// ---------------------------------------------------------------------------
// kernel 2: xW = embed[targets] @ Wi + bi   (M=4096,K=512,N=3072)
// block tile 128x64, 256 threads (8 waves), each wave 2x2 WMMA tiles.
// ---------------------------------------------------------------------------
__global__ void k_gemm_embed(const int* __restrict__ targets,
                             const float* __restrict__ embed,
                             const bf16_t* __restrict__ WiT,   // [3072][512]
                             const float* __restrict__ bi,
                             float* __restrict__ xW) {         // [4096][3072]
    __shared__ __align__(16) bf16_t sA[128 * 32];
    __shared__ __align__(16) bf16_t sB[64 * 32];
    const int m0 = blockIdx.y * 128;
    const int n0 = blockIdx.x * 64;
    const int tid = threadIdx.x;
    const int wave = tid >> 5, wm = wave >> 1, wn = wave & 1;
    const int arow = tid >> 1;                 // 0..127
    const int acol = (tid & 1) * 16;           // 0 or 16
    const float* asrc = embed + (size_t)targets[m0 + arow] * EE + acol;
    v8f acc[2][2] = {};
    for (int k0 = 0; k0 < EE; k0 += 32) {
        { // stage A: gather + fp32->bf16
            const float* s = asrc + k0;
            v8bf lo, hi;
#pragma unroll
            for (int i = 0; i < 8; ++i) { lo[i] = (bf16_t)s[i]; hi[i] = (bf16_t)s[8 + i]; }
            *(v8bf*)&sA[arow * 32 + acol]     = lo;
            *(v8bf*)&sA[arow * 32 + acol + 8] = hi;
        }
#if HAVE_TDM
        if (tid < 32) { // wave0: async tensor DMA of the 64x32 B panel
            tdm_load_2d(lds_off(sB), WiT + (size_t)n0 * EE + k0,
                        32u, 64u, (uint32_t)EE, (uint32_t)(EE - k0), (uint32_t)(GG - n0));
            __builtin_amdgcn_s_wait_tensorcnt(0);
        }
#else
        {
            const int r = tid >> 2, c = (tid & 3) * 8;
            *(v8bf*)&sB[r * 32 + c] = *(const v8bf*)&WiT[(size_t)(n0 + r) * EE + k0 + c];
        }
#endif
        __syncthreads();
        v16bf af0 = load_a_frag(&sA[(wm * 32)      * 32], 32);
        v16bf af1 = load_a_frag(&sA[(wm * 32 + 16) * 32], 32);
        v16bf bf0 = load_b_frag(&sB[(wn * 32)      * 32], 32);
        v16bf bf1 = load_b_frag(&sB[(wn * 32 + 16) * 32], 32);
        acc[0][0] = wmma_bf16(af0, bf0, acc[0][0]);
        acc[0][1] = wmma_bf16(af0, bf1, acc[0][1]);
        acc[1][0] = wmma_bf16(af1, bf0, acc[1][0]);
        acc[1][1] = wmma_bf16(af1, bf1, acc[1][1]);
        __syncthreads();
    }
    const int lane = tid & 31;
    const int nl = lane & 15, mh = (lane >> 4) * 8;
#pragma unroll
    for (int i = 0; i < 2; ++i)
#pragma unroll
        for (int j = 0; j < 2; ++j) {
            const int ng = n0 + wn * 32 + j * 16 + nl;
            const float bias = bi[ng];
#pragma unroll
            for (int r = 0; r < 8; ++r) {
                const int mg = m0 + wm * 32 + i * 16 + r + mh;
                xW[(size_t)mg * GG + ng] = acc[i][j][r] + bias;
            }
        }
}

// ---------------------------------------------------------------------------
// grid-wide barrier for the persistent GRU scan
// ---------------------------------------------------------------------------
__device__ __forceinline__ void grid_sync(unsigned* cnt, unsigned* gen, unsigned nb) {
    __syncthreads();
    if (threadIdx.x == 0) {
        volatile unsigned* vgen = (volatile unsigned*)gen;
        const unsigned g = *vgen;
        __threadfence();
        const unsigned a = atomicAdd(cnt, 1u);
        if (a == nb - 1u) {
            *cnt = 0u;
            __threadfence();
            atomicAdd(gen, 1u);
        } else {
            while (*vgen == g) { __builtin_amdgcn_s_sleep(1); }
        }
    }
    __syncthreads();
}

// ---------------------------------------------------------------------------
// kernel 3: persistent GRU scan. 16 blocks x 128 threads = 64 waves; each wave
// owns one 16-wide hidden slice j0 and computes its r/z/n columns via WMMA.
// h: [16][1024] f32 (ws); hs written directly as bf16 [4096][1024] for GEMM-out.
// ---------------------------------------------------------------------------
__global__ void k_gru_scan(const float* __restrict__ enc,
                           const float* __restrict__ xW,    // [4096][3072]
                           const bf16_t* __restrict__ WhT,  // [3072][1024]
                           const float* __restrict__ bhn,
                           float* __restrict__ h,           // [16][1024]
                           bf16_t* __restrict__ hsb,        // [4096][1024]
                           unsigned* __restrict__ sync) {
    __shared__ __align__(16) bf16_t sh[BB * HH];  // 32 KB: h as bf16
    const int tid = threadIdx.x;
    const int gth = blockIdx.x * 128 + tid;       // 0..2047
#pragma unroll
    for (int i = 0; i < 8; ++i) {                 // h = encoder_state
        const int idx = gth + i * 2048;
        h[idx] = enc[idx];
    }
    grid_sync(sync, sync + 1, gridDim.x);

    const int wave = tid >> 5, lane = tid & 31;
    const int j0 = (blockIdx.x * 4 + wave) * 16;  // hidden slice
    const int nl = lane & 15, mh = (lane >> 4) * 8;
    const int j = j0 + nl;
    const float bh = bhn[j];

    for (int t = 0; t < TT; ++t) {
        // stage h -> LDS as bf16 (coalesced float4 reads)
        for (int i = 0; i < 32; ++i) {
            const int base = tid * 4 + i * 512;
            const float4 v = *(const float4*)&h[base];
            v4bf w = {(bf16_t)v.x, (bf16_t)v.y, (bf16_t)v.z, (bf16_t)v.w};
            *(v4bf*)&sh[base] = w;
        }
        __syncthreads();

        v8f aR = {}, aZ = {}, aN = {};
        for (int k0 = 0; k0 < HH; k0 += 32) {
            v16bf af = load_a_frag(&sh[k0], HH);
            v16bf br = load_b_frag(&WhT[(size_t)(j0)            * HH + k0], HH);
            v16bf bz = load_b_frag(&WhT[(size_t)(HH + j0)       * HH + k0], HH);
            v16bf bn = load_b_frag(&WhT[(size_t)(2 * HH + j0)   * HH + k0], HH);
            aR = wmma_bf16(af, br, aR);
            aZ = wmma_bf16(af, bz, aZ);
            aN = wmma_bf16(af, bn, aN);
        }
        // gates + state update for this wave's (8 batch rows x 1 col) per lane
#pragma unroll
        for (int r = 0; r < 8; ++r) {
            const int b = r + mh;
            const size_t xb = ((size_t)b * TT + t) * GG + j;
            const float xr = xW[xb], xz = xW[xb + HH], xn = xW[xb + 2 * HH];
            const float rg = sigm(xr + aR[r]);
            const float zg = sigm(xz + aZ[r]);
            const float ng = tanhf(xn + rg * (aN[r] + bh));
            const float ho = h[b * HH + j];
            const float hn = (1.0f - zg) * ng + zg * ho;
            h[b * HH + j] = hn;
            hsb[((size_t)b * TT + t) * HH + j] = (bf16_t)hn;
        }
        grid_sync(sync, sync + 1, gridDim.x);
    }
}

// ---------------------------------------------------------------------------
// kernel 4: logits = hs @ Wo + bo   (M=4096,K=1024,N=32000) — dominant GEMM.
// A (hs) already bf16 row-major -> both panels staged via TDM where available.
// ---------------------------------------------------------------------------
__global__ void k_gemm_out(const bf16_t* __restrict__ hsb,   // [4096][1024]
                           const bf16_t* __restrict__ WoT,   // [32000][1024]
                           const float* __restrict__ bo,
                           float* __restrict__ out) {        // [4096][32000]
    __shared__ __align__(16) bf16_t sA[128 * 32];
    __shared__ __align__(16) bf16_t sB[64 * 32];
    const int m0 = blockIdx.y * 128;
    const int n0 = blockIdx.x * 64;
    const int tid = threadIdx.x;
    const int wave = tid >> 5, wm = wave >> 1, wn = wave & 1;
    v8f acc[2][2] = {};
    for (int k0 = 0; k0 < HH; k0 += 32) {
#if HAVE_TDM
        if (tid < 32) { // wave0: async tensor DMA of both panels
            tdm_load_2d(lds_off(sA), hsb + (size_t)m0 * HH + k0,
                        32u, 128u, (uint32_t)HH, (uint32_t)(HH - k0), (uint32_t)(MM - m0));
            tdm_load_2d(lds_off(sB), WoT + (size_t)n0 * HH + k0,
                        32u, 64u, (uint32_t)HH, (uint32_t)(HH - k0), (uint32_t)(VV - n0));
            __builtin_amdgcn_s_wait_tensorcnt(0);
        }
#else
        {
            const int ar = tid >> 1, ac = (tid & 1) * 16;
            *(v8bf*)&sA[ar * 32 + ac]     = *(const v8bf*)&hsb[(size_t)(m0 + ar) * HH + k0 + ac];
            *(v8bf*)&sA[ar * 32 + ac + 8] = *(const v8bf*)&hsb[(size_t)(m0 + ar) * HH + k0 + ac + 8];
            const int br = tid >> 2, bc = (tid & 3) * 8;
            *(v8bf*)&sB[br * 32 + bc] = *(const v8bf*)&WoT[(size_t)(n0 + br) * HH + k0 + bc];
        }
#endif
        __syncthreads();
        v16bf af0 = load_a_frag(&sA[(wm * 32)      * 32], 32);
        v16bf af1 = load_a_frag(&sA[(wm * 32 + 16) * 32], 32);
        v16bf bf0 = load_b_frag(&sB[(wn * 32)      * 32], 32);
        v16bf bf1 = load_b_frag(&sB[(wn * 32 + 16) * 32], 32);
        acc[0][0] = wmma_bf16(af0, bf0, acc[0][0]);
        acc[0][1] = wmma_bf16(af0, bf1, acc[0][1]);
        acc[1][0] = wmma_bf16(af1, bf0, acc[1][0]);
        acc[1][1] = wmma_bf16(af1, bf1, acc[1][1]);
        __syncthreads();
    }
    const int lane = tid & 31;
    const int nl = lane & 15, mh = (lane >> 4) * 8;
#pragma unroll
    for (int i = 0; i < 2; ++i)
#pragma unroll
        for (int j = 0; j < 2; ++j) {
            const int ng = n0 + wn * 32 + j * 16 + nl;
            const float bias = bo[ng];
#pragma unroll
            for (int r = 0; r < 8; ++r) {
                const int mg = m0 + wm * 32 + i * 16 + r + mh;
                out[(size_t)mg * VV + ng] = acc[i][j][r] + bias;
            }
        }
}

// ---------------------------------------------------------------------------
extern "C" void kernel_launch(void* const* d_in, const int* in_sizes, int n_in,
                              void* d_out, int out_size, void* d_ws, size_t ws_size,
                              hipStream_t stream) {
    (void)in_sizes; (void)n_in; (void)out_size; (void)ws_size;
    const float* enc     = (const float*)d_in[0];
    const int*   targets = (const int*)d_in[1];
    const float* embed   = (const float*)d_in[2];
    const float* Wi      = (const float*)d_in[3];
    const float* Wh      = (const float*)d_in[4];
    const float* bi      = (const float*)d_in[5];
    const float* bhn     = (const float*)d_in[6];
    const float* Wo      = (const float*)d_in[7];
    const float* bo      = (const float*)d_in[8];
    float* out = (float*)d_out;

    char* ws = (char*)d_ws;
    float*    h    = (float*)(ws + OFF_H);
    float*    xW   = (float*)(ws + OFF_XW);
    bf16_t*   hsb  = (bf16_t*)(ws + OFF_HS);
    bf16_t*   WiT  = (bf16_t*)(ws + OFF_WIT);
    bf16_t*   WhT  = (bf16_t*)(ws + OFF_WHT);
    bf16_t*   WoT  = (bf16_t*)(ws + OFF_WOT);
    unsigned* sync = (unsigned*)(ws + OFF_SYNC);

    k_init_sync<<<1, 32, 0, stream>>>(sync);
    k_transpose_bf16<<<dim3(GG / 32, EE / 32), dim3(32, 8), 0, stream>>>(Wi, WiT, EE, GG);
    k_transpose_bf16<<<dim3(GG / 32, HH / 32), dim3(32, 8), 0, stream>>>(Wh, WhT, HH, GG);
    k_transpose_bf16<<<dim3(VV / 32, HH / 32), dim3(32, 8), 0, stream>>>(Wo, WoT, HH, VV);
    k_gemm_embed<<<dim3(GG / 64, MM / 128), 256, 0, stream>>>(targets, embed, WiT, bi, xW);
    k_gru_scan<<<16, 128, 0, stream>>>(enc, xW, WhT, bhn, h, hsb, sync);
    k_gemm_out<<<dim3(VV / 64, MM / 128), 256, 0, stream>>>(hsb, WoT, bo, out);
}
